// ValueCurlNet_41661182771590
// MI455X (gfx1250) — matrix operand order repaired and checked by reading, not compiled, not run on hardware
//
#include <hip/hip_runtime.h>
#include <hip/hip_bf16.h>
#include <math.h>

// ---------------- problem constants ----------------
#define BATCH   8192
#define DLAT    128
#define ACT     64
#define DIN     192            // DLAT + ACT
#define HDIM    2048
#define NUPPER  8128           // 128*127/2
#define EPSN    1e-6f

typedef __attribute__((ext_vector_type(16))) __bf16 v16bf;
typedef __attribute__((ext_vector_type(8)))  __bf16 v8bf;
typedef __attribute__((ext_vector_type(8)))  float  v8f;

// ---------------- block reduction helper ----------------
__device__ __forceinline__ float block_reduce_sum(float val, float* red) {
  #pragma unroll
  for (int m = 1; m < 32; m <<= 1) val += __shfl_xor(val, m, 32);
  int lane = threadIdx.x & 31, wid = threadIdx.x >> 5;
  if (lane == 0) red[wid] = val;
  __syncthreads();
  int nw = (blockDim.x + 31) >> 5;
  float r = (threadIdx.x < (unsigned)nw) ? red[threadIdx.x] : 0.0f;
  if (wid == 0) {
    #pragma unroll
    for (int m = 1; m < 32; m <<= 1) r += __shfl_xor(r, m, 32);
    if (lane == 0) red[0] = r;
  }
  __syncthreads();
  float out = red[0];
  __syncthreads();
  return out;
}

// ---------------- init: u2, sigma2 accumulator, (i,j) LUT ----------------
__global__ void k_init(float* sig, float* u2, int* lut) {
  int t = blockIdx.x * blockDim.x + threadIdx.x;
  if (t == 0) sig[1] = 0.0f;
  if (t < NUPPER) {
    u2[t] = 1.0f / sqrtf((float)NUPPER);
    int rem = t, i = 0;
    while (rem >= (DLAT - 1) - i) { rem -= (DLAT - 1) - i; ++i; }
    lut[2 * t] = i;
    lut[2 * t + 1] = i + 1 + rem;
  }
}

// ---------------- W1 spectral norm: single persistent block ----------------
__global__ __launch_bounds__(1024) void k_spectral_w1(const float* __restrict__ W1, float* sig) {
  __shared__ float u[HDIM];
  __shared__ float v[DIN];
  __shared__ float red[32];
  int t = threadIdx.x;
  float inv0 = 1.0f / sqrtf((float)HDIM);
  for (int o = t; o < HDIM; o += 1024) u[o] = inv0;
  __syncthreads();
  for (int it = 0; it < 5; ++it) {
    // v = W1^T u
    float s = 0.0f;
    if (t < DIN) {
      for (int o = 0; o < HDIM; ++o) s += W1[(size_t)o * DIN + t] * u[o];
    }
    __syncthreads();
    if (t < DIN) v[t] = s;
    __syncthreads();
    float ssv = (t < DIN) ? v[t] * v[t] : 0.0f;
    float nv = sqrtf(block_reduce_sum(ssv, red));
    if (t < DIN) v[t] = v[t] / (nv + EPSN);
    __syncthreads();
    // u = W1 v
    float ssu = 0.0f;
    for (int o = t; o < HDIM; o += 1024) {
      float a = 0.0f;
      for (int i = 0; i < DIN; ++i) a += W1[(size_t)o * DIN + i] * v[i];
      u[o] = a; ssu += a * a;
    }
    float nu = sqrtf(block_reduce_sum(ssu, red));
    for (int o = t; o < HDIM; o += 1024) u[o] = u[o] / (nu + EPSN);
    __syncthreads();
  }
  // sigma = u . (W1 v)
  float ps = 0.0f;
  for (int o = t; o < HDIM; o += 1024) {
    float a = 0.0f;
    for (int i = 0; i < DIN; ++i) a += W1[(size_t)o * DIN + i] * v[i];
    ps += u[o] * a;
  }
  float sigma = block_reduce_sum(ps, red);
  if (t == 0) sig[0] = sigma;
}

// ---------------- W2 power iteration step: v2 = W2^T normalize(u2) ----------------
__global__ __launch_bounds__(256) void k_w2_vstep(const float* __restrict__ W2,
                                                  const float* __restrict__ u2,
                                                  float* __restrict__ v2) {
  __shared__ float red[8];
  __shared__ float part[4][64];
  int tid = threadIdx.x;
  float ss = 0.0f;
  for (int o = tid; o < NUPPER; o += 256) { float x = u2[o]; ss += x * x; }
  float nu = sqrtf(block_reduce_sum(ss, red));
  float inv = 1.0f / (nu + EPSN);
  int col = blockIdx.x * 64 + (tid & 63);
  int og = tid >> 6;
  float s = 0.0f;
  for (int o = og; o < NUPPER; o += 4) s += W2[(size_t)o * HDIM + col] * u2[o];
  part[og][tid & 63] = s;
  __syncthreads();
  if (og == 0) {
    float tot = part[0][tid] + part[1][tid] + part[2][tid] + part[3][tid];
    v2[col] = tot * inv;
  }
}

// ---------------- W2 power iteration step: u2 = W2 normalize(v2) ----------------
__global__ __launch_bounds__(256) void k_w2_ustep(const float* __restrict__ W2,
                                                  const float* __restrict__ v2,
                                                  float* __restrict__ u2) {
  __shared__ float red[8];
  int tid = threadIdx.x;
  float ss = 0.0f;
  for (int i = tid; i < HDIM; i += 256) { float x = v2[i]; ss += x * x; }
  float nv = sqrtf(block_reduce_sum(ss, red));
  float inv = 1.0f / (nv + EPSN);
  int w = tid >> 5, lane = tid & 31;
  int o = blockIdx.x * 8 + w;
  float s = 0.0f;
  for (int i = lane; i < HDIM; i += 32) s += W2[(size_t)o * HDIM + i] * v2[i];
  #pragma unroll
  for (int m = 1; m < 32; m <<= 1) s += __shfl_xor(s, m, 32);
  if (lane == 0) u2[o] = s * inv;
}

// ---------------- sigma2 = normalize(u2) . W2 . normalize(v2) ----------------
__global__ __launch_bounds__(256) void k_w2_sigma(const float* __restrict__ W2,
                                                   const float* __restrict__ u2,
                                                   const float* __restrict__ v2,
                                                   float* sig) {
  __shared__ float red[8];
  int tid = threadIdx.x;
  float ssu = 0.0f;
  for (int o = tid; o < NUPPER; o += 256) { float x = u2[o]; ssu += x * x; }
  float nu = sqrtf(block_reduce_sum(ssu, red));
  float ssv = 0.0f;
  for (int i = tid; i < HDIM; i += 256) { float x = v2[i]; ssv += x * x; }
  float nv = sqrtf(block_reduce_sum(ssv, red));
  float invu = 1.0f / (nu + EPSN), invv = 1.0f / (nv + EPSN);
  int w = tid >> 5, lane = tid & 31;
  int o = blockIdx.x * 8 + w;
  float s = 0.0f;
  for (int i = lane; i < HDIM; i += 32) s += W2[(size_t)o * HDIM + i] * v2[i];
  #pragma unroll
  for (int m = 1; m < 32; m <<= 1) s += __shfl_xor(s, m, 32);
  if (lane == 0) atomicAdd(&sig[1], (u2[o] * invu) * (s * invv));
}

// ---------------- f32 -> bf16 conversion ----------------
__global__ void k_f32_to_bf16(const float* __restrict__ src, __hip_bfloat16* __restrict__ dst, int n) {
  int stride = gridDim.x * blockDim.x;
  for (int t = blockIdx.x * blockDim.x + threadIdx.x; t < n; t += stride)
    dst[t] = __float2bfloat16(src[t]);
}

// ---------------- build concat(z, action) in bf16 ----------------
__global__ void k_build_xcat(const float* __restrict__ z, const float* __restrict__ act,
                             __hip_bfloat16* __restrict__ Xc) {
  int t = blockIdx.x * blockDim.x + threadIdx.x;
  if (t >= BATCH * DIN) return;
  int b = t / DIN, c = t % DIN;
  float v = (c < DLAT) ? z[(size_t)b * DLAT + c] : act[(size_t)b * ACT + (c - DLAT)];
  Xc[t] = __float2bfloat16(v);
}

// ---------------- zero diagonal of F ----------------
__global__ void k_diag(float* __restrict__ F) {
  int t = blockIdx.x * blockDim.x + threadIdx.x;
  if (t < BATCH * DLAT) {
    int b = t >> 7, i = t & 127;
    F[((size_t)b * DLAT + i) * DLAT + i] = 0.0f;
  }
}

// ---------------- WMMA fragment loaders (bf16, 16x32 A / 32x16 B) ----------------
// A (MxK): lane m holds row m; half-waves hold K chunks 0..15 / 16..31
__device__ __forceinline__ v16bf load_a_frag(const __hip_bfloat16* base, int row0, int kdim,
                                             int k, int lane) {
  int lm = lane & 15;
  int kc = k + ((lane >> 4) << 4);
  const __hip_bfloat16* p = base + (size_t)(row0 + lm) * kdim + kc;
  v8bf lo = *reinterpret_cast<const v8bf*>(p);
  v8bf hi = *reinterpret_cast<const v8bf*>(p + 8);
  return __builtin_shufflevector(lo, hi, 0,1,2,3,4,5,6,7,8,9,10,11,12,13,14,15);
}
// B (KxN) from row-major B^T (NxK): lane n holds 16 contiguous K; halves split K 0-15/16-31
__device__ __forceinline__ v16bf load_b_frag(const __hip_bfloat16* base, int row0, int kdim,
                                             int k, int lane) {
  int lm = lane & 15;
  int kc = k + ((lane >> 4) << 4);
  return *reinterpret_cast<const v16bf*>(base + (size_t)(row0 + lm) * kdim + kc);
}

#define WMMA_BF16(A, B, C) \
  __builtin_amdgcn_wmma_f32_16x16x32_bf16(false, (A), false, (B), (short)0, (C), false, false)

// ---------------- GEMM1 + fused norm-gated GELU, bf16 out ----------------
// h = Xc @ W1^T / sigma1 + b1 ; gate per 16-bundle ; Hg = bf16(h * gate)
__global__ __launch_bounds__(256) void k_gemm1_gate(const __hip_bfloat16* __restrict__ Xc,
                                                     const __hip_bfloat16* __restrict__ W1b,
                                                     const float* __restrict__ b1,
                                                     const float* __restrict__ sig,
                                                     __hip_bfloat16* __restrict__ Hg) {
  int lane = threadIdx.x & 31;
  int wid = threadIdx.x >> 5;
  int wm = wid >> 1, wn = wid & 1;               // 4x2 wave grid
  int m0 = blockIdx.x * 128 + wm * 32;           // batch rows
  int n0 = blockIdx.y * 64 + wn * 32;            // H columns
  v8f acc[2][2] = {};
  for (int k = 0; k < DIN; k += 32) {
    v16bf a0 = load_a_frag(Xc, m0,      DIN, k, lane);
    v16bf a1 = load_a_frag(Xc, m0 + 16, DIN, k, lane);
    v16bf bb0 = load_b_frag(W1b, n0,      DIN, k, lane);
    v16bf bb1 = load_b_frag(W1b, n0 + 16, DIN, k, lane);
    acc[0][0] = WMMA_BF16(a0, bb0, acc[0][0]);
    acc[0][1] = WMMA_BF16(a0, bb1, acc[0][1]);
    acc[1][0] = WMMA_BF16(a1, bb0, acc[1][0]);
    acc[1][1] = WMMA_BF16(a1, bb1, acc[1][1]);
  }
  float inv_s = 1.0f / sig[0];
  int lm = lane & 15, lh = lane >> 4;
  #pragma unroll
  for (int j = 0; j < 2; ++j) {
    int n = n0 + 16 * j + lm;
    float bias = b1[n];
    #pragma unroll
    for (int i = 0; i < 2; ++i) {
      float h[8], ss[8];
      #pragma unroll
      for (int r = 0; r < 8; ++r) { h[r] = acc[i][j][r] * inv_s + bias; ss[r] = h[r] * h[r]; }
      // bundle (16 consecutive N) sum-of-squares: reduce over the 16-lane half
      #pragma unroll
      for (int m = 1; m < 16; m <<= 1) {
        #pragma unroll
        for (int r = 0; r < 8; ++r) ss[r] += __shfl_xor(ss[r], m, 32);
      }
      #pragma unroll
      for (int r = 0; r < 8; ++r) {
        float nrm = sqrtf(ss[r]);
        float g = 0.5f * nrm * (1.0f + tanhf(0.7978845608028654f * (nrm + 0.044715f * nrm * nrm * nrm)));
        float scale = g / (nrm + EPSN);
        int mrow = m0 + 16 * i + r + 8 * lh;
        Hg[(size_t)mrow * HDIM + n] = __float2bfloat16(h[r] * scale);
      }
    }
  }
}

// ---------------- GEMM2 + fused antisymmetric scatter into F ----------------
// upper = Hg @ W2^T / sigma2 + b2 ; F[b,i,j] = upper, F[b,j,i] = -upper
__global__ __launch_bounds__(256) void k_gemm2_scatter(const __hip_bfloat16* __restrict__ Hg,
                                                        const __hip_bfloat16* __restrict__ W2b,
                                                        const float* __restrict__ b2,
                                                        const float* __restrict__ sig,
                                                        const int* __restrict__ lut,
                                                        float* __restrict__ F) {
  int lane = threadIdx.x & 31;
  int wid = threadIdx.x >> 5;
  int wm = wid >> 1, wn = wid & 1;               // 4x2 wave grid
  int m0 = blockIdx.x * 128 + wm * 32;           // batch rows
  int n0 = blockIdx.y * 64 + wn * 32;            // n_upper columns
  v8f acc[2][2] = {};
  for (int k = 0; k < HDIM; k += 32) {
    v16bf a0 = load_a_frag(Hg, m0,      HDIM, k, lane);
    v16bf a1 = load_a_frag(Hg, m0 + 16, HDIM, k, lane);
    v16bf bb0 = load_b_frag(W2b, n0,      HDIM, k, lane);
    v16bf bb1 = load_b_frag(W2b, n0 + 16, HDIM, k, lane);
    acc[0][0] = WMMA_BF16(a0, bb0, acc[0][0]);
    acc[0][1] = WMMA_BF16(a0, bb1, acc[0][1]);
    acc[1][0] = WMMA_BF16(a1, bb0, acc[1][0]);
    acc[1][1] = WMMA_BF16(a1, bb1, acc[1][1]);
  }
  float inv_s = 1.0f / sig[1];
  int lm = lane & 15, lh = lane >> 4;
  #pragma unroll
  for (int j = 0; j < 2; ++j) {
    int p = n0 + 16 * j + lm;
    float bias = b2[p];
    int iu = lut[2 * p], ju = lut[2 * p + 1];
    #pragma unroll
    for (int i = 0; i < 2; ++i) {
      #pragma unroll
      for (int r = 0; r < 8; ++r) {
        float val = acc[i][j][r] * inv_s + bias;
        int mrow = m0 + 16 * i + r + 8 * lh;
        F[((size_t)mrow * DLAT + iu) * DLAT + ju] = val;
        F[((size_t)mrow * DLAT + ju) * DLAT + iu] = -val;
      }
    }
  }
}

// ---------------- launch ----------------
extern "C" void kernel_launch(void* const* d_in, const int* in_sizes, int n_in,
                              void* d_out, int out_size, void* d_ws, size_t ws_size,
                              hipStream_t stream) {
  const float* z   = (const float*)d_in[0];
  const float* act = (const float*)d_in[1];
  const float* W1  = (const float*)d_in[2];
  const float* b1  = (const float*)d_in[3];
  const float* W2  = (const float*)d_in[4];
  const float* b2  = (const float*)d_in[5];
  float* F = (float*)d_out;

  // workspace carve (all 32B-aligned) — ~71 MB total
  float* sig = (float*)d_ws;                         // [0]=sigma1, [1]=sigma2 (+pad to 16)
  float* u2  = sig + 16;                             // 8128 (reserve 8192)
  float* v2  = u2 + 8192;                            // 2048
  int*   lut = (int*)(v2 + 2048);                    // 2*8128 (reserve 16384)
  __hip_bfloat16* Xc  = (__hip_bfloat16*)(lut + 16384);          // 8192*192
  __hip_bfloat16* W1b = Xc  + (size_t)BATCH * DIN;               // 2048*192
  __hip_bfloat16* W2b = W1b + (size_t)HDIM * DIN;                // 8128*2048
  __hip_bfloat16* Hg  = W2b + (size_t)NUPPER * HDIM;             // 8192*2048

  // init u2 / sigma2 accumulator / (i,j) lookup table
  k_init<<<32, 256, 0, stream>>>(sig, u2, lut);
  // W1 spectral norm (single persistent block)
  k_spectral_w1<<<1, 1024, 0, stream>>>(W1, sig);
  // W2 power iteration (5x), normalization folded into consumers
  for (int it = 0; it < 5; ++it) {
    k_w2_vstep<<<HDIM / 64, 256, 0, stream>>>(W2, u2, v2);
    k_w2_ustep<<<NUPPER / 8, 256, 0, stream>>>(W2, v2, u2);
  }
  k_w2_sigma<<<NUPPER / 8, 256, 0, stream>>>(W2, u2, v2, sig);
  // bf16 conversions
  k_f32_to_bf16<<<4096, 256, 0, stream>>>(W2, W2b, NUPPER * HDIM);
  k_f32_to_bf16<<<512, 256, 0, stream>>>(W1, W1b, HDIM * DIN);
  k_build_xcat<<<(BATCH * DIN + 255) / 256, 256, 0, stream>>>(z, act, Xc);
  // GEMM1 (+ fused gate) : [8192,192] x [192,2048]
  k_gemm1_gate<<<dim3(BATCH / 128, HDIM / 64), 256, 0, stream>>>(Xc, W1b, b1, sig, Hg);
  // zero diagonal, then GEMM2 (+ fused antisymmetric scatter) : [8192,2048] x [2048,8128]
  k_diag<<<(BATCH * DLAT + 255) / 256, 256, 0, stream>>>(F);
  k_gemm2_scatter<<<dim3(BATCH / 128, NUPPER / 64), 256, 0, stream>>>(Hg, W2b, b2, sig, lut, F);
}
